// SelfAttention_91010357002685
// MI455X (gfx1250) — matrix operand order repaired and checked by reading, not compiled, not run on hardware
//
#include <hip/hip_runtime.h>
#include <hip/hip_bf16.h>

// ---------------------------------------------------------------------------
// CDNA5 (gfx1250) self-attention forward, bf16 WMMA path.
// ~155 GFLOP total, working set < 192MB L2 => compute-bound: every GEMM-shaped
// op runs on v_wmma_f32_16x16x32_bf16 (fp32 accumulate). GEMM uses 128x128
// block tiles, double-buffered LDS, and async global->LDS staging (ASYNCcnt)
// where the toolchain exposes it.
// ---------------------------------------------------------------------------

typedef __attribute__((ext_vector_type(16))) __bf16 v16bf;
typedef __attribute__((ext_vector_type(8)))  float  v8f;
typedef __attribute__((ext_vector_type(4)))  int    v4i;

union FragB { unsigned int u[8]; v16bf v; };
union FragF { float f[8]; v8f v; };

#if __has_builtin(__builtin_amdgcn_global_load_async_to_lds_b128)
#define HAVE_ASYNC_LDS 1
#else
#define HAVE_ASYNC_LDS 0
#endif

typedef __attribute__((address_space(1))) v4i gv4i;   // global int4
typedef __attribute__((address_space(3))) v4i lv4i;   // LDS int4

// 16-byte global -> LDS copy; async (ASYNCcnt) when available.
__device__ __forceinline__ void copy_b128(const unsigned short* g, unsigned short* l) {
#if HAVE_ASYNC_LDS
  __builtin_amdgcn_global_load_async_to_lds_b128(
      (gv4i*)(unsigned long long)g,
      (lv4i*)(unsigned int)(unsigned long long)l,   // low 32 bits = LDS offset
      0, 0);
#else
  *(uint4*)l = *(const uint4*)g;
#endif
}

__device__ __forceinline__ void wait_async0() {
#if HAVE_ASYNC_LDS
#if __has_builtin(__builtin_amdgcn_s_wait_asynccnt)
  __builtin_amdgcn_s_wait_asynccnt(0);
#else
  asm volatile("s_wait_asynccnt 0" ::: "memory");
#endif
#endif
}

__device__ __forceinline__ unsigned short f2bf(float x) {
  unsigned int u = __float_as_uint(x);
  unsigned int r = (u + 0x7FFFu + ((u >> 16) & 1u)) >> 16;   // RNE
  return (unsigned short)r;
}

// --------------------------- fp32 -> bf16 convert ---------------------------
__global__ void f32_to_bf16_kernel(const float* __restrict__ src,
                                   unsigned short* __restrict__ dst, int n) {
  int i = blockIdx.x * blockDim.x + threadIdx.x;
  if (i < n) dst[i] = f2bf(src[i]);
}

// ------------------------------- WMMA GEMM ----------------------------------
// C[M,N](fp32) = A[M,K](bf16 rm) * B[K,N](bf16 rm).
// 256 threads = 8 waves; block tile 128x128; wave tile 32x64 (2x4 WMMA tiles,
// 8 WMMAs per 32-wide k-step). Double-buffered LDS:
//   As[buf]: 128x32 row-major (A-frag K-pairs contiguous)  -- async staged
//   Bt[buf]: 128x32 = B transposed [n][k] (B-frag K-pairs contiguous)
//            -- software-pipelined through VGPRs (transpose on store)
__global__ __launch_bounds__(256)
void gemm_bf16_kernel(const unsigned short* __restrict__ A,
                      const unsigned short* __restrict__ Bw,
                      float* __restrict__ C, int M, int N, int K) {
  __shared__ unsigned short As[2][128 * 32];
  __shared__ unsigned short Bt[2][128 * 32];

  const int tid  = threadIdx.x;
  const int lane = tid & 31;
  const int l16  = lane & 15;
  const int half = lane >> 4;
  const int wid  = tid >> 5;
  const int wr   = wid >> 1;   // 0..3 -> m offset 32*wr
  const int wc   = wid & 1;    // 0..1 -> n offset 64*wc
  const int bm   = blockIdx.y * 128;
  const int bn   = blockIdx.x * 128;

  FragF acc[2][4];
#pragma unroll
  for (int i = 0; i < 2; i++)
#pragma unroll
    for (int j = 0; j < 4; j++)
#pragma unroll
      for (int r = 0; r < 8; r++) acc[i][j].f[r] = 0.0f;

  // A staging: 128x32 tile = 512 b128 chunks; 2 per thread.
  const int bkk = tid & 31;          // B: k within tile
  const int bn0 = (tid >> 5) * 8;    // B: n start (0..56), + 64 for 2nd chunk

  auto stageA = [&](int kt, int buf) {
    const unsigned short* base = A + (size_t)bm * K + kt * 32;
#pragma unroll
    for (int c = 0; c < 2; c++) {
      const int id  = tid + c * 256;
      const int row = id >> 2;
      const int c8  = (id & 3) * 8;
      copy_b128(base + (size_t)row * K + c8, &As[buf][row * 32 + c8]);
    }
  };
  auto loadB = [&](int kt, uint4* bv) {
    const unsigned short* base = Bw + (size_t)(kt * 32 + bkk) * N + bn + bn0;
    bv[0] = *(const uint4*)(base);
    bv[1] = *(const uint4*)(base + 64);
  };
  auto scatterB = [&](const uint4* bv, int buf) {
#pragma unroll
    for (int c = 0; c < 2; c++) {
      const unsigned short* be = (const unsigned short*)&bv[c];
#pragma unroll
      for (int e = 0; e < 8; e++)
        Bt[buf][(bn0 + c * 64 + e) * 32 + bkk] = be[e];
    }
  };

  const int nT = K >> 5;
  uint4 bv[2];
  stageA(0, 0);
  loadB(0, bv);
  scatterB(bv, 0);
  wait_async0();
  __syncthreads();

  for (int kt = 0; kt < nT; kt++) {
    const int cur = kt & 1, nxt = cur ^ 1;
    const bool more = (kt + 1) < nT;
    if (more) {
      stageA(kt + 1, nxt);    // async: no VGPR round-trip
      loadB(kt + 1, bv);      // B in flight while we compute
    }

    FragB af[2], bf4[4];
#pragma unroll
    for (int mt = 0; mt < 2; mt++) {
      const int row = wr * 32 + mt * 16 + l16;
#pragma unroll
      for (int v = 0; v < 8; v++) {
        const int kk = 2 * v + 16 * half;  // A frag layout
        af[mt].u[v] = *(const unsigned int*)(&As[cur][row * 32 + kk]);
      }
    }
#pragma unroll
    for (int nt = 0; nt < 4; nt++) {
      const int col = wc * 64 + nt * 16 + l16;
#pragma unroll
      for (int v = 0; v < 8; v++) {
        const int kk = 2 * v + 16 * half;                      // B frag layout
        bf4[nt].u[v] = *(const unsigned int*)(&Bt[cur][col * 32 + kk]);
      }
    }
#pragma unroll
    for (int mt = 0; mt < 2; mt++)
#pragma unroll
      for (int nt = 0; nt < 4; nt++)
        acc[mt][nt].v = __builtin_amdgcn_wmma_f32_16x16x32_bf16(
            false, af[mt].v, false, bf4[nt].v, (short)0, acc[mt][nt].v,
            false, false);

    if (more) {
      scatterB(bv, nxt);
      wait_async0();
    }
    __syncthreads();
  }

  // C/D layout: lanes 0-15: VGPR r -> M=r; lanes 16-31: VGPR r -> M=8+r
#pragma unroll
  for (int mt = 0; mt < 2; mt++) {
#pragma unroll
    for (int r = 0; r < 8; r++) {
      const int row = bm + wr * 32 + mt * 16 + r + 8 * half;
#pragma unroll
      for (int nt = 0; nt < 4; nt++) {
        C[(size_t)row * N + bn + wc * 64 + nt * 16 + l16] = acc[mt][nt].f[r];
      }
    }
  }
}

// ------------------------------- RoPE ---------------------------------------
// src: fp32 [B][S][nh*128]; dst: bf16 [B][nh][S][128]; scale folded in (Q gets
// 1/sqrt(128); rotation is orthogonal so pre-scaling commutes).
__global__ void rope_kernel(const float* __restrict__ src,
                            const float* __restrict__ fcos,
                            const float* __restrict__ fsin,
                            unsigned short* __restrict__ dst,
                            int nheads, float scale, int total) {
  int i = blockIdx.x * blockDim.x + threadIdx.x;
  if (i >= total) return;
  int p = i & 63;             // rotary pair index (0..63)
  int rest = i >> 6;          // (b*S+s)*nh + h
  int h = rest % nheads;
  int bs = rest / nheads;     // b*S + s
  int s = bs & 2047;
  int b = bs >> 11;
  float c = fcos[s * 64 + p];
  float sn = fsin[s * 64 + p];
  size_t si = ((size_t)bs * nheads + h) * 128 + 2 * p;
  float x1 = src[si], x2 = src[si + 1];
  float o1 = (x1 * c - x2 * sn) * scale;
  float o2 = (x1 * sn + x2 * c) * scale;
  size_t di = (((size_t)b * nheads + h) * 2048 + s) * 128 + 2 * p;
  dst[di] = f2bf(o1);
  dst[di + 1] = f2bf(o2);
}

// V: fp32 [B][S][4*128] -> bf16 [B][4][S][128]
__global__ void v_convert_kernel(const float* __restrict__ src,
                                 unsigned short* __restrict__ dst, int total) {
  int i = blockIdx.x * blockDim.x + threadIdx.x;
  if (i >= total) return;
  int d = i & 127;
  int t = i >> 7;
  int h = t & 3;
  int bs = t >> 2;            // b*S + s
  int s = bs & 2047;
  int b = bs >> 11;
  dst[(((size_t)b * 4 + h) * 2048 + s) * 128 + d] = f2bf(src[i]);
}

// --------------------------- Flash attention --------------------------------
// Block = 128 threads (4 waves); each wave owns 16 q rows, block covers 64.
// kv processed in tiles of 32 with online softmax; causal mask analytic.
__global__ __launch_bounds__(128)
void attn_kernel(const unsigned short* __restrict__ Q,   // [B][16][S][128]
                 const unsigned short* __restrict__ Km,  // [B][4][S][128]
                 const unsigned short* __restrict__ Vm,  // [B][4][S][128]
                 unsigned short* __restrict__ O) {       // [B][S][2048]
  __shared__ unsigned short Vt[128 * 32];       // V tile transposed [d][kv]
  __shared__ unsigned short Pb[4 * 16 * 32];    // per-wave P tile [16][32]

  const int S = 2048;
  const int qt = blockIdx.x & 31;         // S/64 = 32 q tiles
  const int h  = (blockIdx.x >> 5) & 15;
  const int b  = blockIdx.x >> 9;

  const int tid = threadIdx.x;
  const int lane = tid & 31, l16 = lane & 15, half = lane >> 4;
  const int wid = tid >> 5;
  const int q0b = qt * 64;
  const int q0  = q0b + wid * 16;
  const int kvh = h >> 2;                 // n_rep = 4

  const unsigned short* qp = Q  + ((size_t)b * 16 + h)   * S * 128;
  const unsigned short* kp = Km + ((size_t)b * 4  + kvh) * S * 128;
  const unsigned short* vp = Vm + ((size_t)b * 4  + kvh) * S * 128;

  // Q fragments for this wave's 16 rows: 4 k-steps of 32 over head_dim=128
  FragB qf[4];
#pragma unroll
  for (int kc = 0; kc < 4; kc++) {
    const unsigned short* qrow = qp + (size_t)(q0 + l16) * 128 + kc * 32;
#pragma unroll
    for (int v = 0; v < 8; v++) {
      const int kk = 2 * v + 16 * half;
      qf[kc].u[v] = *(const unsigned int*)(qrow + kk);
    }
  }

  FragF o[8];
#pragma unroll
  for (int n = 0; n < 8; n++)
#pragma unroll
    for (int r = 0; r < 8; r++) o[n].f[r] = 0.0f;
  float mrow[8], lrow[8];
#pragma unroll
  for (int r = 0; r < 8; r++) { mrow[r] = -1e30f; lrow[r] = 0.0f; }

  const int nIter = (q0b + 64) >> 5;      // kv tiles of 32, causal bound
  for (int j = 0; j < nIter; j++) {
    const int kvb = j * 32;

    // Stage V tile transposed: thread -> kv = tid&31, d chunk = (tid>>5)*32
    {
      const int kv = tid & 31;
      const int d0 = (tid >> 5) * 32;
      const unsigned short* vr = vp + (size_t)(kvb + kv) * 128 + d0;
#pragma unroll
      for (int c = 0; c < 4; c++) {
        uint4 vv = *(const uint4*)(vr + c * 8);
        const unsigned short* ve = (const unsigned short*)&vv;
#pragma unroll
        for (int e = 0; e < 8; e++) Vt[(d0 + c * 8 + e) * 32 + kv] = ve[e];
      }
    }
    // Prefetch next kv tile of K and V into cache (global_prefetch_b8)
    if (j + 1 < nIter) {
      __builtin_prefetch(kp + (size_t)(kvb + 32 + (tid & 31)) * 128, 0, 1);
      __builtin_prefetch(vp + (size_t)(kvb + 32 + (tid & 31)) * 128, 0, 1);
    }
    __syncthreads();

    if (kvb <= q0 + 15) {
      // ---- scores: S = Q (16x128) * K^T (128x32), two 16x16 tiles ----
      FragF s0, s1;
#pragma unroll
      for (int r = 0; r < 8; r++) { s0.f[r] = 0.0f; s1.f[r] = 0.0f; }
#pragma unroll
      for (int t = 0; t < 2; t++) {
        const int kvcol = kvb + t * 16 + l16;  // B-matrix N = kv row
        FragF* st = t ? &s1 : &s0;
#pragma unroll
        for (int kc = 0; kc < 4; kc++) {
          FragB kf;
#pragma unroll
          for (int v = 0; v < 8; v++) {
            const int kk = 2 * v + 16 * half + kc * 32;  // contiguous d pair
            kf.u[v] = *(const unsigned int*)(kp + (size_t)kvcol * 128 + kk);
          }
          st->v = __builtin_amdgcn_wmma_f32_16x16x32_bf16(
              false, qf[kc].v, false, kf.v, (short)0, st->v, false, false);
        }
      }

      // ---- causal mask + online softmax (rows live across 16-lane halves) --
#pragma unroll
      for (int r = 0; r < 8; r++) {
        const int row = q0 + r + 8 * half;
        const int c0 = kvb + l16;
        if (c0 > row) s0.f[r] = -1e30f;
        if (c0 + 16 > row) s1.f[r] = -1e30f;
        float tm = fmaxf(s0.f[r], s1.f[r]);
#pragma unroll
        for (int m2 = 8; m2 >= 1; m2 >>= 1) tm = fmaxf(tm, __shfl_xor(tm, m2, 32));
        const float nm = fmaxf(mrow[r], tm);
        const float alpha = __expf(mrow[r] - nm);
        const float p0 = __expf(s0.f[r] - nm);
        const float p1 = __expf(s1.f[r] - nm);
        float rs = p0 + p1;
#pragma unroll
        for (int m2 = 8; m2 >= 1; m2 >>= 1) rs += __shfl_xor(rs, m2, 32);
        lrow[r] = lrow[r] * alpha + rs;
        mrow[r] = nm;
#pragma unroll
        for (int n = 0; n < 8; n++) o[n].f[r] *= alpha;
        // P -> LDS (C layout -> row-major [16][32]); same-wave LDS is in-order
        unsigned short* pw = Pb + wid * 16 * 32 + (r + 8 * half) * 32;
        pw[l16] = f2bf(p0);
        pw[16 + l16] = f2bf(p1);
      }

      // Re-read P as an A-fragment (short-typed reads: no TBAA reordering)
      FragB pf;
      const unsigned short* pr = Pb + wid * 16 * 32 + l16 * 32;
#pragma unroll
      for (int v = 0; v < 8; v++) {
        const int kk = 2 * v + 16 * half;
        pf.u[v] = (unsigned int)pr[kk] | ((unsigned int)pr[kk + 1] << 16);
      }

      // ---- O += P (16x32) * V (32x128): 8 WMMAs over d tiles ----
#pragma unroll
      for (int n = 0; n < 8; n++) {
        FragB vf;
#pragma unroll
        for (int v = 0; v < 8; v++) {
          const int kk = 2 * v + 16 * half;  // kv pair, contiguous in Vt row
          vf.u[v] = *(const unsigned int*)(Vt + (n * 16 + l16) * 32 + kk);
        }
        o[n].v = __builtin_amdgcn_wmma_f32_16x16x32_bf16(
            false, pf.v, false, vf.v, (short)0, o[n].v, false, false);
      }
    }
    __syncthreads();
  }

  // ---- epilogue: normalize and write bf16 [B][S][h*128 + d] ----
  const int hbase = h * 128;
#pragma unroll
  for (int r = 0; r < 8; r++) {
    const int row = q0 + r + 8 * half;
    const float inv = 1.0f / lrow[r];
    unsigned short* orow = O + ((size_t)b * S + row) * 2048 + hbase;
#pragma unroll
    for (int n = 0; n < 8; n++) orow[n * 16 + l16] = f2bf(o[n].f[r] * inv);
  }
}

// ------------------------------- launch -------------------------------------
extern "C" void kernel_launch(void* const* d_in, const int* in_sizes, int n_in,
                              void* d_out, int out_size, void* d_ws, size_t ws_size,
                              hipStream_t stream) {
  (void)in_sizes; (void)n_in; (void)out_size; (void)ws_size;
  constexpr int B = 2, S = 2048, DIM = 2048, H = 16, KVH = 4;
  constexpr int M = B * S;             // 4096
  constexpr int KVD = KVH * 128;       // 512

  const float* x    = (const float*)d_in[0];
  const float* fcos = (const float*)d_in[1];
  const float* fsin = (const float*)d_in[2];
  // d_in[3] = mask (causal, handled analytically)
  const float* wq   = (const float*)d_in[4];
  const float* wk   = (const float*)d_in[5];
  const float* wv   = (const float*)d_in[6];
  const float* wo   = (const float*)d_in[7];

  char* ws = (char*)d_ws;
  size_t off = 0;
  auto alloc = [&](size_t bytes) -> void* {
    void* p = ws + off;
    off += (bytes + 255) & ~(size_t)255;
    return p;
  };
  unsigned short* xb    = (unsigned short*)alloc((size_t)M * DIM * 2);
  unsigned short* wqb   = (unsigned short*)alloc((size_t)DIM * DIM * 2);
  unsigned short* wkb   = (unsigned short*)alloc((size_t)DIM * KVD * 2);
  unsigned short* wvb   = (unsigned short*)alloc((size_t)DIM * KVD * 2);
  unsigned short* wob   = (unsigned short*)alloc((size_t)DIM * DIM * 2);
  float*          qf    = (float*)alloc((size_t)M * DIM * 4);
  float*          kf    = (float*)alloc((size_t)M * KVD * 4);
  float*          vf    = (float*)alloc((size_t)M * KVD * 4);
  unsigned short* qb    = (unsigned short*)alloc((size_t)M * DIM * 2);
  unsigned short* kb    = (unsigned short*)alloc((size_t)M * KVD * 2);
  unsigned short* vb    = (unsigned short*)alloc((size_t)M * KVD * 2);
  unsigned short* attnb = (unsigned short*)alloc((size_t)M * DIM * 2);

  auto cvt = [&](const float* s, unsigned short* d, int n) {
    f32_to_bf16_kernel<<<(n + 255) / 256, 256, 0, stream>>>(s, d, n);
  };
  cvt(x,  xb,  M * DIM);
  cvt(wq, wqb, DIM * DIM);
  cvt(wk, wkb, DIM * KVD);
  cvt(wv, wvb, DIM * KVD);
  cvt(wo, wob, DIM * DIM);

  // Projections: fp32 accumulate via bf16 WMMA (128x128 block tiles)
  gemm_bf16_kernel<<<dim3(DIM / 128, M / 128), 256, 0, stream>>>(xb, wqb, qf, M, DIM, DIM);
  gemm_bf16_kernel<<<dim3(KVD / 128, M / 128), 256, 0, stream>>>(xb, wkb, kf, M, KVD, DIM);
  gemm_bf16_kernel<<<dim3(KVD / 128, M / 128), 256, 0, stream>>>(xb, wvb, vf, M, KVD, DIM);

  // RoPE + layout transform to [b][h][s][d]; 1/sqrt(128) folded into Q
  {
    int totq = B * S * H * 64;
    rope_kernel<<<(totq + 255) / 256, 256, 0, stream>>>(
        qf, fcos, fsin, qb, H, 0.08838834764831845f, totq);
    int totk = B * S * KVH * 64;
    rope_kernel<<<(totk + 255) / 256, 256, 0, stream>>>(
        kf, fcos, fsin, kb, KVH, 1.0f, totk);
    int totv = B * S * KVD;
    v_convert_kernel<<<(totv + 255) / 256, 256, 0, stream>>>(vf, vb, totv);
  }

  // Flash attention: grid = B * H * (S/64)
  attn_kernel<<<B * H * (S / 64), 128, 0, stream>>>(qb, kb, vb, attnb);

  // Output projection into fp32 d_out
  gemm_bf16_kernel<<<dim3(DIM / 128, M / 128), 256, 0, stream>>>(
      attnb, wob, (float*)d_out, M, DIM, DIM);
}
